// GPTLanguageModel_10943576670407
// MI455X (gfx1250) — compile-verified, matrix operand order checked by reading, not run on hardware
//
#include <hip/hip_runtime.h>
#include <stdint.h>

// ---------------------------------------------------------------------------
// GPT forward (6 layers, B=256, T=256, C=384) for gfx1250 (MI455X, wave32).
// GEMMs + attention einsums on v_wmma_f32_16x16x32_f16 (f16 in, f32 acc).
// GEMM A-tiles staged via Tensor Data Mover (tensor_load_to_lds +
// s_wait_tensorcnt) when the toolchain exposes the builtin; double-buffered
// LDS, one barrier per K-step.
// ---------------------------------------------------------------------------

#define VOCABSZ 27
#define NEMBD   384
#define SEQT    256
#define NHEADS  6
#define HEADD   64
#define NLAYERS 6
#define NBATCH  256
#define BT      (NBATCH * SEQT)   // 65536 tokens

typedef __attribute__((ext_vector_type(16))) _Float16 v16h;
typedef __attribute__((ext_vector_type(8)))  _Float16 v8h;
typedef __attribute__((ext_vector_type(8)))  float    v8f;
typedef __attribute__((ext_vector_type(4)))  unsigned int v4u;
typedef __attribute__((ext_vector_type(8)))  int v8i;
typedef __attribute__((ext_vector_type(4)))  int v4i;

#if defined(__has_builtin)
#if __has_builtin(__builtin_amdgcn_tensor_load_to_lds) && \
    __has_builtin(__builtin_amdgcn_s_wait_tensorcnt)
#define USE_TDM 1
#endif
#endif
#ifndef USE_TDM
#define USE_TDM 0
#endif

// Build a 16x32 f16 A/B fragment (wave32 layout): per lane, two contiguous
// 8-half runs (K = kb..kb+7 and K = 16+kb..16+kb+7), each a 16-byte load.
__device__ __forceinline__ v16h make_frag(const _Float16* p0, const _Float16* p1) {
  union { v16h w; v8h h[2]; } u;
  u.h[0] = *(const v8h*)p0;
  u.h[1] = *(const v8h*)p1;
  return u.w;
}

__device__ __forceinline__ v8f wmma16(v16h a, v16h b, v8f c) {
  // (neg_a, A, neg_b, B, c_mod, C, reuse_a, reuse_b)
  return __builtin_amdgcn_wmma_f32_16x16x32_f16(false, a, false, b, (short)0, c,
                                                false, false);
}

#if USE_TDM
// Issue a TDM 2D tile load (16-bit elements) into LDS.
//   tensor_w/h : tensor extent (elements) for OOB checks (tiles are in-bounds)
//   stride     : row stride in elements
//   tile_w/h   : tile extent (elements); tile_w contiguous in memory
// D# bit layout per CDNA5 ISA ch.8 (group0 128b, group1 256b).
// This toolchain's builtin arity: (v4u g0, v8i g1, v4i g2, v4i g3, v8i, i32 cpol).
__device__ __forceinline__ void tdm_load_2d(uint32_t lds_off, const void* gp,
                                            uint32_t tensor_w, uint32_t tensor_h,
                                            uint32_t stride, uint32_t tile_w,
                                            uint32_t tile_h) {
  uint64_t ga = (uint64_t)(uintptr_t)gp;
  v4u g0 = {
      1u,                                               // count=1 (valid D#)
      lds_off,                                          // lds_addr (bytes)
      (uint32_t)ga,                                     // global_addr[31:0]
      (uint32_t)((ga >> 32) & 0x1FFFFFFu) | (2u << 30)  // addr[56:32] | type=2
  };
  v8i g1 = {
      (int)(1u << 16),                                     // data_size=1 (2B)
      (int)((tensor_w & 0xFFFFu) << 16),                   // td0[15:0] @63:48
      (int)((tensor_w >> 16) | ((tensor_h & 0xFFFFu) << 16)),  // td0[31:16], td1[15:0]
      (int)(((tensor_h >> 16) & 0xFFFFu) | (tile_w << 16)),    // td1[31:16], tile_dim0
      (int)tile_h,                                         // tile_dim1 (tile_dim2=0)
      (int)stride,                                         // td0_stride[31:0]
      0, 0};
  v4i z4 = {0, 0, 0, 0};
  v8i z8 = {0, 0, 0, 0, 0, 0, 0, 0};
  __builtin_amdgcn_tensor_load_to_lds(g0, g1, z4, z4, z8, 0);
}
#endif

// ---------------------------------------------------------------------------
// Tiled WMMA GEMM: C[M,N] = A[M,K] * B[K,N] (+bias) (+residual) (relu?)
// Block tile 64x128, K-step 32, 8 waves in 2(m) x 4(n); wave tile 32x32 =
// 4 WMMAs per K-step. Double-buffered LDS, one barrier per K-step.
// A tile staged by TDM (wave 0) or coalesced b128 loads; B tile staged
// transposed ([n][k]) so all fragment reads are contiguous ds_load_b128.
// ---------------------------------------------------------------------------
template <bool BIAS, bool RELU, bool RES, bool OUT16, bool OUT32>
__global__ __launch_bounds__(256) void gemm_wmma_kernel(
    const _Float16* __restrict__ A, int lda,
    const _Float16* __restrict__ Bm, int ldb,
    const float* __restrict__ bias,
    const float* __restrict__ res,
    _Float16* __restrict__ out16,
    float* __restrict__ out32,
    int M, int N, int K) {
  __shared__ _Float16 As[2][64 * 32];    // [m][k]
  __shared__ _Float16 Bs[2][128 * 32];   // [n][k] (transposed)

  const int tid  = threadIdx.x;
  const int lane = tid & 31;
  const int wave = tid >> 5;
  const int wm   = wave & 1;        // 0..1 -> 32-row block
  const int wn   = wave >> 1;       // 0..3 -> 32-col block
  const int m0   = blockIdx.y * 64;
  const int n0   = blockIdx.x * 128;

  const int arow = tid >> 2;        // 0..63
  const int acol = (tid & 3) * 8;   // 0,8,16,24
  const int brow = tid >> 3;        // 0..31 (k)
  const int bcol = (tid & 7) * 16;  // 0..112 (n)

  const int fr = lane & 15;
  const int fk = (lane >> 4) * 8;

  v8f acc[2][2] = {};
  const int nk = K >> 5;

  auto stage = [&](int buf, int k0) {
#if USE_TDM
    if (wave == 0) {
      tdm_load_2d((uint32_t)(uintptr_t)&As[buf][0],
                  &A[(size_t)m0 * lda + k0],
                  (uint32_t)K, (uint32_t)M, (uint32_t)lda, 32u, 64u);
    }
#else
    *(v8h*)&As[buf][arow * 32 + acol] =
        *(const v8h*)&A[(size_t)(m0 + arow) * lda + k0 + acol];
#endif
    v8h b0 = *(const v8h*)&Bm[(size_t)(k0 + brow) * ldb + n0 + bcol];
    v8h b1 = *(const v8h*)&Bm[(size_t)(k0 + brow) * ldb + n0 + bcol + 8];
    if (k0 + 32 < K) {  // global_prefetch_b8 for the tile after next
      __builtin_prefetch(&Bm[(size_t)(k0 + 32 + brow) * ldb + n0 + bcol], 0, 1);
    }
#pragma unroll
    for (int i = 0; i < 8; ++i) Bs[buf][(bcol + i) * 32 + brow] = b0[i];
#pragma unroll
    for (int i = 0; i < 8; ++i) Bs[buf][(bcol + 8 + i) * 32 + brow] = b1[i];
  };

  auto settle = [&]() {
#if USE_TDM
    if (wave == 0) __builtin_amdgcn_s_wait_tensorcnt(0);
#endif
    __syncthreads();
  };

  auto compute = [&](int buf) {
    v16h af0 = make_frag(&As[buf][(wm * 32 + fr) * 32 + fk],
                         &As[buf][(wm * 32 + fr) * 32 + 16 + fk]);
    v16h af1 = make_frag(&As[buf][(wm * 32 + 16 + fr) * 32 + fk],
                         &As[buf][(wm * 32 + 16 + fr) * 32 + 16 + fk]);
    v16h bf0 = make_frag(&Bs[buf][(wn * 32 + fr) * 32 + fk],
                         &Bs[buf][(wn * 32 + fr) * 32 + 16 + fk]);
    v16h bf1 = make_frag(&Bs[buf][(wn * 32 + 16 + fr) * 32 + fk],
                         &Bs[buf][(wn * 32 + 16 + fr) * 32 + 16 + fk]);
    acc[0][0] = wmma16(af0, bf0, acc[0][0]);
    acc[0][1] = wmma16(af0, bf1, acc[0][1]);
    acc[1][0] = wmma16(af1, bf0, acc[1][0]);
    acc[1][1] = wmma16(af1, bf1, acc[1][1]);
  };

  stage(0, 0);
  settle();
  int cur = 0;
  for (int t = 1; t < nk; ++t) {
    stage(cur ^ 1, t << 5);
    compute(cur);
    settle();
    cur ^= 1;
  }
  compute(cur);

  // Epilogue. C/D layout: VGPR r -> M = r + 8*(lane>=16), N = lane&15.
  const int half = lane >> 4;
  const int col  = lane & 15;
#pragma unroll
  for (int mi = 0; mi < 2; ++mi) {
#pragma unroll
    for (int ni = 0; ni < 2; ++ni) {
      const int gn = n0 + wn * 32 + ni * 16 + col;
      const float bvs = BIAS ? bias[gn] : 0.0f;
#pragma unroll
      for (int r = 0; r < 8; ++r) {
        const int gm = m0 + wm * 32 + mi * 16 + 8 * half + r;
        float v = acc[mi][ni][r] + bvs;
        if (RES)  v += res[(size_t)gm * N + gn];
        if (RELU) v = v > 0.0f ? v : 0.0f;
        if (OUT32) out32[(size_t)gm * N + gn] = v;
        if (OUT16) out16[(size_t)gm * N + gn] = (_Float16)v;
      }
    }
  }
}

// ---------------------------------------------------------------------------
// Attention: one wave per (batch*head, 16-row q block).
// S = (Q K^T)/8 with causal mask -> softmax -> Y = P V, both on WMMA.
// qkv rows: [q(0:384) | k(384:768) | v(768:1152)], head h at offset h*64.
// ---------------------------------------------------------------------------
__global__ __launch_bounds__(32) void attention_kernel(
    const _Float16* __restrict__ qkv,   // [BT, 1152]
    _Float16* __restrict__ y) {         // [BT, 384]
  __shared__ float    S[16 * 256];      // scores (f32)
  __shared__ _Float16 P[16 * 256];      // softmax probs (f16)
  __shared__ _Float16 Vt[64 * 256];     // V transposed: [n][k]

  const int lane = threadIdx.x;
  const int bh = blockIdx.x;            // 0..B*NHEADS-1
  const int qb = blockIdx.y;            // 0..T/16-1
  const int b  = bh / NHEADS;
  const int h  = bh % NHEADS;
  const size_t base = (size_t)b * SEQT * 1152;

  // Stage V^T into LDS (coalesced along n).
  for (int idx = lane; idx < 64 * 256; idx += 32) {
    const int k = idx >> 6;
    const int n = idx & 63;
    Vt[n * 256 + k] = qkv[base + (size_t)k * 1152 + 768 + h * 64 + n];
  }
  __syncthreads();

  const int fr = lane & 15;
  const int fk = (lane >> 4) * 8;

  // Q fragments for this 16-row block (K=64 split into two 32-chunks).
  const _Float16* qrow = &qkv[base + (size_t)(qb * 16 + fr) * 1152 + h * 64];
  v16h aq0 = make_frag(qrow + fk,      qrow + 16 + fk);
  v16h aq1 = make_frag(qrow + 32 + fk, qrow + 48 + fk);

  // Scores: 16 key tiles of 16 columns each.
  for (int j = 0; j < 16; ++j) {
    // B = K^T, so B[k][n] = K_keys[j*16+n][k] -> contiguous in head dim.
    const _Float16* krow =
        &qkv[base + (size_t)(j * 16 + fr) * 1152 + 384 + h * 64];
    v16h bk0 = make_frag(krow + fk,      krow + 16 + fk);
    v16h bk1 = make_frag(krow + 32 + fk, krow + 48 + fk);
    v8f s = {};
    s = wmma16(aq0, bk0, s);
    s = wmma16(aq1, bk1, s);
    const int scol = j * 16 + fr;
#pragma unroll
    for (int rr = 0; rr < 8; ++rr) {
      const int srow = rr + 8 * (lane >> 4);
      float v = s[rr] * 0.125f;                 // 1/sqrt(64)
      if (scol > qb * 16 + srow) v = -3.0e38f;  // causal mask
      S[srow * 256 + scol] = v;
    }
  }
  __syncthreads();

  // Softmax: two lanes per row (each handles 128 cols), combine via shfl_xor.
  {
    const int row  = lane & 15;
    const int half = lane >> 4;
    float* sp = &S[row * 256 + half * 128];
    float mx = -3.4e38f;
    for (int c = 0; c < 128; ++c) mx = fmaxf(mx, sp[c]);
    mx = fmaxf(mx, __shfl_xor(mx, 16, 32));
    float sum = 0.0f;
    for (int c = 0; c < 128; ++c) {
      float e = __expf(sp[c] - mx);
      sp[c] = e;
      sum += e;
    }
    sum += __shfl_xor(sum, 16, 32);
    const float inv = 1.0f / sum;
    _Float16* pp = &P[row * 256 + half * 128];
    for (int c = 0; c < 128; ++c) pp[c] = (_Float16)(sp[c] * inv);
  }
  __syncthreads();

  // Y = P[16,256] @ V[256,64]: 4 n-tiles x 8 K-chunks of WMMA.
  for (int nt = 0; nt < 4; ++nt) {
    v8f acc = {};
    for (int kc = 0; kc < 256; kc += 32) {
      v16h ap = make_frag(&P[fr * 256 + kc + fk], &P[fr * 256 + kc + 16 + fk]);
      v16h bv = make_frag(&Vt[(nt * 16 + fr) * 256 + kc + fk],
                          &Vt[(nt * 16 + fr) * 256 + kc + 16 + fk]);
      acc = wmma16(ap, bv, acc);
    }
#pragma unroll
    for (int rr = 0; rr < 8; ++rr) {
      const int row = rr + 8 * (lane >> 4);
      y[(size_t)(b * SEQT + qb * 16 + row) * NEMBD + h * 64 + nt * 16 +
        (lane & 15)] = (_Float16)acc[rr];
    }
  }
}

// ---------------------------------------------------------------------------
// LayerNorm: one wave per row of 384; shfl reductions; writes f32 and/or f16.
// ---------------------------------------------------------------------------
__global__ __launch_bounds__(256) void layernorm_kernel(
    const float* __restrict__ in, const float* __restrict__ g,
    const float* __restrict__ bt, float* __restrict__ out32,
    _Float16* __restrict__ out16, int C) {
  const int row  = blockIdx.x * 8 + (threadIdx.x >> 5);
  const int lane = threadIdx.x & 31;
  const float* p = in + (size_t)row * C;
  float s = 0.0f;
  for (int c = lane; c < C; c += 32) s += p[c];
#pragma unroll
  for (int o = 16; o; o >>= 1) s += __shfl_xor(s, o, 32);
  const float mu = s / (float)C;
  float v = 0.0f;
  for (int c = lane; c < C; c += 32) {
    float d = p[c] - mu;
    v += d * d;
  }
#pragma unroll
  for (int o = 16; o; o >>= 1) v += __shfl_xor(v, o, 32);
  const float rstd = rsqrtf(v / (float)C + 1e-5f);
  for (int c = lane; c < C; c += 32) {
    float o = (p[c] - mu) * rstd * g[c] + bt[c];
    if (out32) out32[(size_t)row * C + c] = o;
    if (out16) out16[(size_t)row * C + c] = (_Float16)o;
  }
}

__global__ void embed_kernel(const int* __restrict__ x,
                             const float* __restrict__ tok,
                             const float* __restrict__ pos,
                             float* __restrict__ h, _Float16* __restrict__ hh,
                             int total) {
  int i = blockIdx.x * blockDim.x + threadIdx.x;
  if (i >= total) return;
  int row = i / NEMBD, c = i % NEMBD;
  int t = row % SEQT;
  float v = tok[(size_t)x[row] * NEMBD + c] + pos[(size_t)t * NEMBD + c];
  h[i] = v;
  hh[i] = (_Float16)v;
}

__global__ void cvt_f32_f16_kernel(const float* __restrict__ in,
                                   _Float16* __restrict__ out, int n) {
  int i = blockIdx.x * blockDim.x + threadIdx.x;
  if (i < n) out[i] = (_Float16)in[i];
}

__global__ void logits_kernel(const _Float16* __restrict__ hh,
                              const float* __restrict__ w,
                              const float* __restrict__ b,
                              float* __restrict__ out, int total) {
  int i = blockIdx.x * blockDim.x + threadIdx.x;
  if (i >= total) return;
  int row = i / VOCABSZ, col = i % VOCABSZ;
  const _Float16* hp = hh + (size_t)row * NEMBD;
  float s = b[col];
  for (int k = 0; k < NEMBD; ++k) s += (float)hp[k] * w[(size_t)k * VOCABSZ + col];
  out[i] = s;
}

__global__ void zero_loss_kernel(float* p) {
  if (blockIdx.x == 0 && threadIdx.x == 0) *p = 0.0f;
}

__global__ void loss_kernel(const float* __restrict__ logits,
                            const int* __restrict__ tgt,
                            float* __restrict__ loss, int rows) {
  int row = blockIdx.x * blockDim.x + threadIdx.x;
  if (row >= rows) return;
  const float* lp = logits + (size_t)row * VOCABSZ;
  float mx = lp[0];
  for (int c = 1; c < VOCABSZ; ++c) mx = fmaxf(mx, lp[c]);
  float s = 0.0f;
  for (int c = 0; c < VOCABSZ; ++c) s += __expf(lp[c] - mx);
  float l = lp[tgt[row]] - mx - __logf(s);
  atomicAdd(loss, -l / (float)rows);
}

// ---------------------------------------------------------------------------
// Host orchestration. Input order = setup_inputs() dict insertion order:
//   0:x 1:targets 2:tok_emb 3:pos_emb
//   4+11*l+{0..10}: attn_w, proj_w, proj_b, ff_w1, ff_b1, ff_w2, ff_b2,
//                   ln1_g, ln1_b, ln2_g, ln2_b   (l = 0..5)
//   70:ln_g 71:ln_b 72:lm_w 73:lm_b
// Workspace use ~676 MB.
// ---------------------------------------------------------------------------
extern "C" void kernel_launch(void* const* d_in, const int* in_sizes, int n_in,
                              void* d_out, int out_size, void* d_ws,
                              size_t ws_size, hipStream_t stream) {
  (void)in_sizes; (void)n_in; (void)out_size; (void)ws_size;
  const int*   x      = (const int*)d_in[0];
  const int*   tgt    = (const int*)d_in[1];
  const float* tok    = (const float*)d_in[2];
  const float* pos    = (const float*)d_in[3];
  const float* lnf_g  = (const float*)d_in[70];
  const float* lnf_b  = (const float*)d_in[71];
  const float* lm_w   = (const float*)d_in[72];
  const float* lm_b   = (const float*)d_in[73];

  uint8_t* wsp = (uint8_t*)d_ws;
  auto carve = [&](size_t bytes) -> void* {
    void* p = (void*)wsp;
    wsp += (bytes + 255) & ~(size_t)255;
    return p;
  };

  _Float16* hh   = (_Float16*)carve((size_t)BT * NEMBD * 2);      // f16 stream
  float*    h    = (float*)carve((size_t)BT * NEMBD * 4);         // f32 residual
  float*    r    = (float*)carve((size_t)BT * NEMBD * 4);         // pre-LN sum
  _Float16* qkv  = (_Float16*)carve((size_t)BT * 3 * NEMBD * 2);
  _Float16* yb   = (_Float16*)carve((size_t)BT * NEMBD * 2);
  _Float16* ub   = (_Float16*)carve((size_t)BT * 4 * NEMBD * 2);
  _Float16* wq16[NLAYERS];
  _Float16* wp16[NLAYERS];
  _Float16* w116[NLAYERS];
  _Float16* w216[NLAYERS];
  for (int l = 0; l < NLAYERS; ++l) {
    wq16[l] = (_Float16*)carve((size_t)NEMBD * 3 * NEMBD * 2);
    wp16[l] = (_Float16*)carve((size_t)NEMBD * NEMBD * 2);
    w116[l] = (_Float16*)carve((size_t)NEMBD * 4 * NEMBD * 2);
    w216[l] = (_Float16*)carve((size_t)4 * NEMBD * NEMBD * 2);
  }

  auto cvt = [&](const float* src, _Float16* dst, int n) {
    cvt_f32_f16_kernel<<<(n + 255) / 256, 256, 0, stream>>>(src, dst, n);
  };
  for (int l = 0; l < NLAYERS; ++l) {
    cvt((const float*)d_in[4 + 11 * l + 0], wq16[l], NEMBD * 3 * NEMBD);
    cvt((const float*)d_in[4 + 11 * l + 1], wp16[l], NEMBD * NEMBD);
    cvt((const float*)d_in[4 + 11 * l + 3], w116[l], NEMBD * 4 * NEMBD);
    cvt((const float*)d_in[4 + 11 * l + 5], w216[l], 4 * NEMBD * NEMBD);
  }

  // Embedding
  {
    int total = BT * NEMBD;
    embed_kernel<<<(total + 255) / 256, 256, 0, stream>>>(x, tok, pos, h, hh,
                                                          total);
  }

  for (int l = 0; l < NLAYERS; ++l) {
    const float* proj_b = (const float*)d_in[4 + 11 * l + 2];
    const float* ff_b1  = (const float*)d_in[4 + 11 * l + 4];
    const float* ff_b2  = (const float*)d_in[4 + 11 * l + 6];
    const float* ln1_g  = (const float*)d_in[4 + 11 * l + 7];
    const float* ln1_b  = (const float*)d_in[4 + 11 * l + 8];
    const float* ln2_g  = (const float*)d_in[4 + 11 * l + 9];
    const float* ln2_b  = (const float*)d_in[4 + 11 * l + 10];

    // qkv = hh @ Wqkv   [BT,1152]
    gemm_wmma_kernel<false, false, false, true, false>
        <<<dim3(3 * NEMBD / 128, BT / 64), 256, 0, stream>>>(
            hh, NEMBD, wq16[l], 3 * NEMBD, nullptr, nullptr, qkv, nullptr, BT,
            3 * NEMBD, NEMBD);

    // attention -> yb [BT,384]
    attention_kernel<<<dim3(NBATCH * NHEADS, SEQT / 16), 32, 0, stream>>>(qkv,
                                                                          yb);

    // r = h + yb @ Wproj + b
    gemm_wmma_kernel<true, false, true, false, true>
        <<<dim3(NEMBD / 128, BT / 64), 256, 0, stream>>>(
            yb, NEMBD, wp16[l], NEMBD, proj_b, h, nullptr, r, BT, NEMBD, NEMBD);
    layernorm_kernel<<<BT / 8, 256, 0, stream>>>(r, ln1_g, ln1_b, h, hh, NEMBD);

    // ub = relu(hh @ W1 + b1)   [BT,1536]
    gemm_wmma_kernel<true, true, false, true, false>
        <<<dim3(4 * NEMBD / 128, BT / 64), 256, 0, stream>>>(
            hh, NEMBD, w116[l], 4 * NEMBD, ff_b1, nullptr, ub, nullptr, BT,
            4 * NEMBD, NEMBD);

    // r = h + ub @ W2 + b2
    gemm_wmma_kernel<true, false, true, false, true>
        <<<dim3(NEMBD / 128, BT / 64), 256, 0, stream>>>(
            ub, 4 * NEMBD, w216[l], NEMBD, ff_b2, h, nullptr, r, BT, NEMBD,
            4 * NEMBD);
    layernorm_kernel<<<BT / 8, 256, 0, stream>>>(r, ln2_g, ln2_b, h, hh, NEMBD);
  }

  // Final LN -> hh only (feeds LM head)
  layernorm_kernel<<<BT / 8, 256, 0, stream>>>(h, lnf_g, lnf_b, nullptr, hh,
                                               NEMBD);

  // Logits -> d_out[0 .. BT*V), loss -> d_out[BT*V]
  float* out = (float*)d_out;
  {
    int total = BT * VOCABSZ;
    logits_kernel<<<(total + 255) / 256, 256, 0, stream>>>(hh, lm_w, lm_b, out,
                                                           total);
  }
  float* loss = out + (size_t)BT * VOCABSZ;
  zero_loss_kernel<<<1, 32, 0, stream>>>(loss);
  loss_kernel<<<BT / 256, 256, 0, stream>>>(out, tgt, loss, BT);
}
